// SANLayer_10582799417881
// MI455X (gfx1250) — matrix operand order repaired and checked by reading, not compiled, not run on hardware
//
#include <hip/hip_runtime.h>
#include <hip/hip_bf16.h>

typedef __attribute__((ext_vector_type(16))) _Float16 v16h;
typedef __attribute__((ext_vector_type(8)))  _Float16 v8h;
typedef __attribute__((ext_vector_type(4)))  _Float16 v4h;
typedef __attribute__((ext_vector_type(8)))  float    v8f;

#define NN 50000
#define EE 800000
#define DD 128
#define NEG_SLOPE 0.2f

// ---------- float <-> order-preserving uint key (for single-op atomic max) ----------
__device__ __forceinline__ unsigned fkey(float f) {
    unsigned u = __float_as_uint(f);
    return (u & 0x80000000u) ? ~u : (u | 0x80000000u);
}
__device__ __forceinline__ float funkey(unsigned k) {
    unsigned u = (k & 0x80000000u) ? (k & 0x7FFFFFFFu) : ~k;
    return __uint_as_float(u);
}
#define KEY_NEG_INF 0x007FFFFFu   // fkey(-inf)

// ---------- prep: f16 staging, W transpose, softmax-state init, out zero ----------
__global__ __launch_bounds__(256) void san_prep_kernel(
    const float* __restrict__ X,
    const float* __restrict__ Wu, const float* __restrict__ Wd, const float* __restrict__ Wp,
    _Float16* __restrict__ Xh, _Float16* __restrict__ Wt,
    unsigned* __restrict__ mkey_u, unsigned* __restrict__ mkey_d,
    float* __restrict__ denom_u, float* __restrict__ denom_d,
    float* __restrict__ out)
{
    int idx = blockIdx.x * 256 + threadIdx.x;           // grid covers NN*DD
    if (idx < NN * DD) {
        Xh[idx] = (_Float16)X[idx];
        out[idx] = 0.0f;
    }
    if (idx < 3 * DD * DD) {                            // Wt[branch][n][k] = W[k][n]
        int br  = idx / (DD * DD);
        int rem = idx - br * (DD * DD);
        int n = rem / DD, k = rem - (rem / DD) * DD;
        const float* W = (br == 0) ? Wu : ((br == 1) ? Wd : Wp);
        Wt[idx] = (_Float16)W[k * DD + n];
    }
    if (idx < NN) {
        mkey_u[idx] = KEY_NEG_INF;  mkey_d[idx] = KEY_NEG_INF;
        denom_u[idx] = 0.0f;        denom_d[idx] = 0.0f;
    }
}

// ---------- fused 3-branch GEMM: H[br] = (f16)(Xh @ W[br] + b[br]) ----------
// block = 128 threads (4 waves). Each block: one 16-row strip; each wave: 32 cols.
// A operand: lanes 0-15 hold M=0..15 with K 0..15;
// lanes 16-31 hold K 16..31. B from transposed W: 16 contiguous f16 per lane.
__global__ __launch_bounds__(128) void san_gemm_kernel(
    const _Float16* __restrict__ Xh, const _Float16* __restrict__ Wt,
    const float* __restrict__ bu, const float* __restrict__ bd, const float* __restrict__ bp,
    _Float16* __restrict__ H)
{
    const int branch = blockIdx.y;
    const _Float16* W  = Wt + (size_t)branch * DD * DD;
    const float*   bias = (branch == 0) ? bu : ((branch == 1) ? bd : bp);
    _Float16* Hb = H + (size_t)branch * NN * DD;

    const int lane = threadIdx.x & 31;
    const int wave = threadIdx.x >> 5;
    const int row0 = blockIdx.x * 16;
    const int colBase = wave * 32;
    const int m15 = lane & 15;
    const int hi  = lane >> 4;        // lane-half selector

    v8f acc0 = {};  v8f acc1 = {};
#pragma unroll
    for (int k0 = 0; k0 < DD; k0 += 32) {
        // A: 16x32 f16 tile
        const _Float16* ap = Xh + (size_t)(row0 + m15) * DD + k0 + hi * 16;
        v8h alo = *reinterpret_cast<const v8h*>(ap);
        v8h ahi = *reinterpret_cast<const v8h*>(ap + 8);
        v16h a;
#pragma unroll
        for (int i = 0; i < 8; ++i) { a[i] = alo[i]; a[8 + i] = ahi[i]; }
        // B: 32x16 f16 tiles from transposed W (16 contiguous halves per lane)
        const _Float16* bq0 = W + (size_t)(colBase + m15) * DD + k0 + hi * 16;
        const _Float16* bq1 = W + (size_t)(colBase + 16 + m15) * DD + k0 + hi * 16;
        v16h b0 = *reinterpret_cast<const v16h*>(bq0);
        v16h b1 = *reinterpret_cast<const v16h*>(bq1);
        acc0 = __builtin_amdgcn_wmma_f32_16x16x32_f16(false, a, false, b0,
                                                      (short)0, acc0, false, false);
        acc1 = __builtin_amdgcn_wmma_f32_16x16x32_f16(false, a, false, b1,
                                                      (short)0, acc1, false, false);
    }
    // C/D layout: lanes 0-15 -> N=lane, M=r ; lanes 16-31 -> N=lane-16, M=8+r
    const int c0 = colBase + m15, c1 = colBase + 16 + m15;
    const float bias0 = bias[c0], bias1 = bias[c1];
#pragma unroll
    for (int r = 0; r < 8; ++r) {
        _Float16* o = Hb + (size_t)(row0 + hi * 8 + r) * DD;
        o[c0] = (_Float16)(acc0[r] + bias0);
        o[c1] = (_Float16)(acc1[r] + bias1);
    }
}

// ---------- per-node attention scalars: s = h . a (both branches) ----------
__global__ __launch_bounds__(256) void san_scores_kernel(
    const _Float16* __restrict__ H,
    const float* __restrict__ asu, const float* __restrict__ adu,
    const float* __restrict__ asd, const float* __restrict__ add_,
    float* __restrict__ ssu, float* __restrict__ sdu,
    float* __restrict__ ssd, float* __restrict__ sdd)
{
    const int lane = threadIdx.x & 31;
    const int node = blockIdx.x * 8 + (threadIdx.x >> 5);
    if (node >= NN) return;
    const _Float16* hu = H + (size_t)node * DD + lane * 4;
    const _Float16* hd = H + (size_t)NN * DD + (size_t)node * DD + lane * 4;
    float p0 = 0, p1 = 0, p2 = 0, p3 = 0;
#pragma unroll
    for (int j = 0; j < 4; ++j) {
        int c = lane * 4 + j;
        float xu = (float)hu[j], xd = (float)hd[j];
        p0 += xu * asu[c];  p1 += xu * adu[c];
        p2 += xd * asd[c];  p3 += xd * add_[c];
    }
#pragma unroll
    for (int off = 16; off > 0; off >>= 1) {
        p0 += __shfl_xor(p0, off, 32);
        p1 += __shfl_xor(p1, off, 32);
        p2 += __shfl_xor(p2, off, 32);
        p3 += __shfl_xor(p3, off, 32);
    }
    if (lane == 0) { ssu[node] = p0; sdu[node] = p1; ssd[node] = p2; sdd[node] = p3; }
}

__device__ __forceinline__ float leaky(float x) {
    return x > 0.0f ? x : NEG_SLOPE * x;
}

// ---------- softmax pass 1: segment max (ordered-key atomicMax) ----------
__global__ __launch_bounds__(256) void san_edge_max_kernel(
    const int* __restrict__ idx, const float* __restrict__ ss,
    const float* __restrict__ sd, unsigned* __restrict__ mkey)
{
    int e = blockIdx.x * 256 + threadIdx.x;
    if (e >= EE) return;
    int row = idx[e], col = idx[EE + e];
    float x = leaky(ss[row] + sd[col]);
    atomicMax(mkey + row, fkey(x));
}

// ---------- softmax pass 2: segment sum of exp ----------
__global__ __launch_bounds__(256) void san_edge_sum_kernel(
    const int* __restrict__ idx, const float* __restrict__ ss,
    const float* __restrict__ sd, const unsigned* __restrict__ mkey,
    float* __restrict__ denom)
{
    int e = blockIdx.x * 256 + threadIdx.x;
    if (e >= EE) return;
    int row = idx[e], col = idx[EE + e];
    float x = leaky(ss[row] + sd[col]);
    atomicAdd(denom + row, expf(x - funkey(mkey[row])));
}

// ---------- aggregation: out[row] += alpha * h[col]   (wave per edge) ----------
__global__ __launch_bounds__(256) void san_edge_agg_kernel(
    const int* __restrict__ idx, const float* __restrict__ ss,
    const float* __restrict__ sd, const unsigned* __restrict__ mkey,
    const float* __restrict__ denom, const _Float16* __restrict__ Hb,
    float* __restrict__ out)
{
    int e = blockIdx.x * 8 + (threadIdx.x >> 5);
    if (e >= EE) return;
    int lane = threadIdx.x & 31;
    int row = idx[e], col = idx[EE + e];
    float x = leaky(ss[row] + sd[col]);
    float alpha = expf(x - funkey(mkey[row])) / (denom[row] + 1e-16f);
    v4h h = *reinterpret_cast<const v4h*>(Hb + (size_t)col * DD + lane * 4);
    float* op = out + (size_t)row * DD + lane * 4;
#pragma unroll
    for (int j = 0; j < 4; ++j) atomicAdd(op + j, alpha * (float)h[j]);
}

// ---------- p branch: out[row] += p_val[e] * h_p[col] ----------
__global__ __launch_bounds__(256) void san_p_agg_kernel(
    const int* __restrict__ idx, const float* __restrict__ pval,
    const _Float16* __restrict__ Hp, float* __restrict__ out)
{
    int e = blockIdx.x * 8 + (threadIdx.x >> 5);
    if (e >= EE) return;
    int lane = threadIdx.x & 31;
    int row = idx[e], col = idx[EE + e];
    float s = pval[e];
    v4h h = *reinterpret_cast<const v4h*>(Hp + (size_t)col * DD + lane * 4);
    float* op = out + (size_t)row * DD + lane * 4;
#pragma unroll
    for (int j = 0; j < 4; ++j) atomicAdd(op + j, s * (float)h[j]);
}

extern "C" void kernel_launch(void* const* d_in, const int* in_sizes, int n_in,
                              void* d_out, int out_size, void* d_ws, size_t ws_size,
                              hipStream_t stream) {
    const float* features = (const float*)d_in[0];
    const int*   lu       = (const int*)d_in[1];
    const int*   ld       = (const int*)d_in[2];
    const int*   pidx     = (const int*)d_in[3];
    const float* pval     = (const float*)d_in[4];
    const float* Wu   = (const float*)d_in[5];
    const float* bu   = (const float*)d_in[6];
    const float* asu  = (const float*)d_in[7];
    const float* adu  = (const float*)d_in[8];
    const float* Wd   = (const float*)d_in[9];
    const float* bd   = (const float*)d_in[10];
    const float* asd  = (const float*)d_in[11];
    const float* add_ = (const float*)d_in[12];
    const float* Wp   = (const float*)d_in[13];
    const float* bp   = (const float*)d_in[14];
    float* out = (float*)d_out;

    // workspace carve-up (~51 MB)
    char* ws = (char*)d_ws;
    _Float16* Xh = (_Float16*)ws;                 ws += (size_t)NN * DD * 2;      // 12.8 MB
    _Float16* Wt = (_Float16*)ws;                 ws += (size_t)3 * DD * DD * 2;  // 96 KB
    _Float16* H  = (_Float16*)ws;                 ws += (size_t)3 * NN * DD * 2;  // 38.4 MB
    float* ssu = (float*)ws;                      ws += (size_t)NN * 4;
    float* sdu = (float*)ws;                      ws += (size_t)NN * 4;
    float* ssd = (float*)ws;                      ws += (size_t)NN * 4;
    float* sdd = (float*)ws;                      ws += (size_t)NN * 4;
    unsigned* mku = (unsigned*)ws;                ws += (size_t)NN * 4;
    unsigned* mkd = (unsigned*)ws;                ws += (size_t)NN * 4;
    float* den_u = (float*)ws;                    ws += (size_t)NN * 4;
    float* den_d = (float*)ws;                    ws += (size_t)NN * 4;

    const _Float16* Hu = H;
    const _Float16* Hd = H + (size_t)NN * DD;
    const _Float16* Hp = H + (size_t)2 * NN * DD;

    // 1) stage f16 inputs, transpose weights, init softmax state, zero out
    san_prep_kernel<<<(NN * DD) / 256, 256, 0, stream>>>(
        features, Wu, Wd, Wp, Xh, Wt, mku, mkd, den_u, den_d, out);

    // 2) three GEMMs via WMMA: grid.x = 50000/16 row tiles, grid.y = branch
    san_gemm_kernel<<<dim3(NN / 16, 3), 128, 0, stream>>>(Xh, Wt, bu, bd, bp, H);

    // 3) attention score scalars (both GAT branches)
    san_scores_kernel<<<NN / 8, 256, 0, stream>>>(
        H, asu, adu, asd, add_, ssu, sdu, ssd, sdd);

    // 4) segment softmax passes
    san_edge_max_kernel<<<EE / 256, 256, 0, stream>>>(lu, ssu, sdu, mku);
    san_edge_max_kernel<<<EE / 256, 256, 0, stream>>>(ld, ssd, sdd, mkd);
    san_edge_sum_kernel<<<EE / 256, 256, 0, stream>>>(lu, ssu, sdu, mku, den_u);
    san_edge_sum_kernel<<<EE / 256, 256, 0, stream>>>(ld, ssd, sdd, mkd, den_d);

    // 5) aggregation (all three branches accumulate into out)
    san_edge_agg_kernel<<<EE / 8, 256, 0, stream>>>(lu, ssu, sdu, mku, den_u, Hu, out);
    san_edge_agg_kernel<<<EE / 8, 256, 0, stream>>>(ld, ssd, sdd, mkd, den_d, Hd, out);
    san_p_agg_kernel<<<EE / 8, 256, 0, stream>>>(pidx, pval, Hp, out);

    (void)in_sizes; (void)n_in; (void)out_size; (void)ws_size;
}
